// UnimoEncoder_28200755265586
// MI455X (gfx1250) — matrix operand order repaired and checked by reading, not compiled, not run on hardware
//
#include <hip/hip_runtime.h>
#include <hip/hip_bf16.h>
#include <cstdint>

typedef __attribute__((ext_vector_type(16))) _Float16 v16h;
typedef __attribute__((ext_vector_type(8)))  _Float16 v8h;
typedef __attribute__((ext_vector_type(8)))  float    v8f;
typedef __attribute__((ext_vector_type(4)))  unsigned v4ui;
typedef __attribute__((ext_vector_type(8)))  int      v8i;
typedef __attribute__((ext_vector_type(4)))  int      v4i;

namespace cfg {
constexpr int B  = 16;
constexpr int S  = 512;
constexpr int H  = 768;
constexpr int NH = 12;
constexpr int HD = 64;
constexpr int FF = 3072;
constexpr int L  = 50;
constexpr int KV = L + S;   // 562 real key length
constexpr int KP = 576;     // padded key length (multiple of 64)
constexpr long MTOK = (long)B * S;  // 8192 token rows
constexpr int BROW = 72;    // LDS B-tile row pitch in halfs (64 data + 8 pad)
}

// ---------------------------------------------------------------------------
// fp32 -> f16 grid-stride convert
// ---------------------------------------------------------------------------
__global__ __launch_bounds__(256)
void cvt_f32_f16(const float* __restrict__ src, _Float16* __restrict__ dst, long n) {
  for (long i = (long)blockIdx.x * blockDim.x + threadIdx.x; i < n;
       i += (long)gridDim.x * blockDim.x)
    dst[i] = (_Float16)src[i];
}

// ---------------------------------------------------------------------------
// LDS-tiled transpose: src fp32 [K,N] row-major -> dst f16 [N,K] row-major
// ---------------------------------------------------------------------------
__global__ __launch_bounds__(256)
void transpose_f32_to_f16(const float* __restrict__ src, _Float16* __restrict__ dst,
                          int K, int N) {
  __shared__ float tile[32][33];
  const int bx = blockIdx.x * 32;           // along N
  const int by = blockIdx.y * 32;           // along K
  const int tx = threadIdx.x & 31;
  const int ty = threadIdx.x >> 5;          // 0..7
  #pragma unroll
  for (int i = 0; i < 32; i += 8) {
    const int k = by + ty + i, n = bx + tx;
    if (k < K && n < N) tile[ty + i][tx] = src[(long)k * N + n];
  }
  __syncthreads();
  #pragma unroll
  for (int i = 0; i < 32; i += 8) {
    const int n = bx + ty + i, k = by + tx;
    if (n < N && k < K) dst[(long)n * K + k] = (_Float16)tile[tx][ty + i];
  }
}

// ---------------------------------------------------------------------------
// past_key fp32 [B,NH,L,HD] -> f16 into k16[bh][0..L), v16[bh][0..L) (source
// bug: past_key goes into BOTH), and transposed vT16[bh][d][0..L)
// ---------------------------------------------------------------------------
__global__ __launch_bounds__(256)
void past_to_kv(const float* __restrict__ past, _Float16* __restrict__ k16,
                _Float16* __restrict__ v16, _Float16* __restrict__ vT16) {
  using namespace cfg;
  const long i = (long)blockIdx.x * blockDim.x + threadIdx.x;
  const long total = (long)B * NH * L * HD;
  if (i >= total) return;
  const int d = (int)(i & (HD - 1));
  const long t = i >> 6;           // bh*L + l
  const int l = (int)(t % L);
  const long bh = t / L;
  const _Float16 hv = (_Float16)past[i];
  k16[(bh * KP + l) * HD + d] = hv;
  v16[(bh * KP + l) * HD + d] = hv;
  vT16[(bh * HD + d) * KP + l] = hv;
}

// zero the padded key rows [KV, KP) so padded GEMM reads are benign
__global__ __launch_bounds__(256)
void zero_pads(_Float16* __restrict__ k16, _Float16* __restrict__ v16,
               _Float16* __restrict__ vT16) {
  using namespace cfg;
  const long i = (long)blockIdx.x * blockDim.x + threadIdx.x;
  const long total = (long)B * NH * (KP - KV) * HD;
  if (i >= total) return;
  const int d = (int)(i & (HD - 1));
  const long t = i >> 6;
  const int r = (int)(t % (KP - KV));
  const long bh = t / (KP - KV);
  const int kk = KV + r;
  k16[(bh * KP + kk) * HD + d] = (_Float16)0.0f;
  v16[(bh * KP + kk) * HD + d] = (_Float16)0.0f;
  vT16[(bh * HD + d) * KP + kk] = (_Float16)0.0f;
}

// ---------------------------------------------------------------------------
// TDM: DMA one 64(row) x 64(half) f16 tile of Bt (row-major, stride ldb)
// into LDS with a 144-byte row pitch (pad 16B per 128B row => bank-conflict-
// free fragment reads). Descriptor per CDNA5 ISA.
//   group0: count=1 | lds_addr | global_addr | type=2
//   group1: data_size=8B, pad_enable, pad_interval=32DW, pad_amount=4DW,
//           tensor_dim0/1 = 1<<20 (tiles always in-bounds),
//           tile_dim0=16 (x8B = 128B row), tile_dim1=64,
//           tensor_dim0_stride = ldb/4 (8B units)
// ---------------------------------------------------------------------------
__device__ inline void tdm_issue_b(const _Float16* gsrc, unsigned lds_off,
                                   unsigned stride_units) {
  const unsigned long long ga = (unsigned long long)(uintptr_t)gsrc;
  v4ui g0;
  g0[0] = 1u;                                   // count=1, user descriptor
  g0[1] = lds_off;                              // lds_addr (bytes)
  g0[2] = (unsigned)ga;                         // global_addr[31:0]
  g0[3] = (unsigned)(ga >> 32) | (2u << 30);    // global_addr[56:32] | type=2
  v8i g1;
  g1[0] = 0x07130000;        // data_size=3(8B) | pad_en | interval=32DW | amt=4DW
  g1[1] = 0;                 // barrier_addr=0, tensor_dim0 low16 (1<<20 -> 0)
  g1[2] = 0x00000010;        // tensor_dim0 hi16=0x10, tensor_dim1 low16=0
  g1[3] = 0x00100010;        // tensor_dim1 hi16=0x10 | tile_dim0=16
  g1[4] = 64;                // tile_dim1=64, tile_dim2=0
  g1[5] = (int)stride_units; // tensor_dim0_stride low32 (8B units)
  g1[6] = 0;                 // stride hi16=0, dim1_stride low16=0
  g1[7] = 0;
  const v4i gz = {0, 0, 0, 0};
#if defined(__clang_major__) && __clang_major__ >= 23
  const v8i gz8 = {0, 0, 0, 0, 0, 0, 0, 0};
  __builtin_amdgcn_tensor_load_to_lds(g0, g1, gz, gz, gz8, 0);
#else
  __builtin_amdgcn_tensor_load_to_lds(g0, g1, gz, gz, 0);
#endif
}

// ---------------------------------------------------------------------------
// WMMA GEMM: C[M,N] = A[M,K] (f16, row-major, lda) x B (given as Bt[N,K] f16,
// row-major, ldb). Wave tile 32x64 (2 M sub-tiles x 4 N sub-tiles = 8 acc),
// 8 waves/block => block tile 256x64, K-chunk 64 halfs. Each B fragment from
// LDS feeds TWO WMMAs (both M sub-tiles) -> half the LDS traffic / dscnt
// stalls per FLOP vs a 16-row tile. The shared 64x64 B tile is TDM-DMA'd
// into LDS (double buffered, one issuing wave per block); A fragments stream
// from global so the compiler pipelines them with LOADcnt.
//
// Fragment layouts per CDNA5 ISA:
//   A: lane(row)=l&15, hi=l>>4; two contiguous 8-half runs per k-step.
//   B: lane(col)=l&15, b[i]=Bt[col, k] in A's order: two 8-half runs at k=hi*8 and 16+hi*8.
//
// Modes: 0=Q-proj 1=K-proj 2=V-proj(+vT) 3=scores(batched) 4=ctx(batched)
//        5=Wo->f32 6=Wi+GELU->f16 7=Wf->f32
// ---------------------------------------------------------------------------
template<int MODE>
__global__ __launch_bounds__(256)
void gemm_wmma(const _Float16* __restrict__ A, const _Float16* __restrict__ Bt,
               const float* __restrict__ bias,
               void* __restrict__ out0, void* __restrict__ out1,
               const float* __restrict__ amask,
               int M, int N, int K, int lda, int ldb) {
  using namespace cfg;
  (void)M; (void)N;
  __shared__ __align__(16) _Float16 bbuf[2][64 * BROW];   // 2 x 9216 B

  const int tid  = threadIdx.x;
  const int lane = tid & 31;
  const int row  = lane & 15;   // also the B-fragment column
  const int hi   = lane >> 4;
  const int wave = tid >> 5;
  // wave id is uniform per wave: make it *compiler-provably* uniform so the
  // TDM issue is under a scalar branch (TDM ignores EXEC -> every wave that
  // reaches it would DMA again).
  const int uwave = __builtin_amdgcn_readfirstlane(wave);

  const int m0 = blockIdx.y * 256 + wave * 32;
  const int n0 = blockIdx.x * 64;

  if constexpr (MODE == 3) {            // scores: per (b,h) slices
    const long bh = blockIdx.z;
    A  += bh * (long)S * HD;            // q   [S,HD]
    Bt += bh * (long)KP * HD;           // k   [KP,HD] == Bt layout
  } else if constexpr (MODE == 4) {     // ctx: per (b,h) slices
    const long bh = blockIdx.z;
    A  += bh * (long)S * KP;            // probs [S,KP]
    Bt += bh * (long)HD * KP;           // vT    [HD,KP] == Bt layout
  }

  const unsigned lds0 = (unsigned)(uintptr_t)&bbuf[0][0];
  const unsigned lds1 = (unsigned)(uintptr_t)&bbuf[1][0];
  const unsigned su   = (unsigned)(ldb >> 2);   // ldb halfs * 2B / 8B
  const _Float16* btile = Bt + (long)n0 * ldb;  // 64 rows starting at n0

  const _Float16* ap0 = A + (long)(m0 + row) * lda + hi * 8;        // M rows 0..15
  const _Float16* ap1 = A + (long)(m0 + 16 + row) * lda + hi * 8;   // M rows 16..31
  v8f acc[2][4] = {};

  const int nchunks = K >> 6;
  if (uwave == 0) {
    tdm_issue_b(btile, lds0, su);
    __builtin_amdgcn_s_wait_tensorcnt(0);
  }
  __syncthreads();

  for (int kc = 0; kc < nchunks; ++kc) {
    if (uwave == 0 && kc + 1 < nchunks)
      tdm_issue_b(btile + (kc + 1) * 64, (kc & 1) ? lds0 : lds1, su);

    const _Float16* lb = bbuf[kc & 1];
    #pragma unroll
    for (int ks = 0; ks < 2; ++ks) {
      const int k0 = kc * 64 + ks * 32;
      const v8h a0lo = *(const v8h*)(ap0 + k0);
      const v8h a0hi = *(const v8h*)(ap0 + k0 + 16);
      const v16h a0 = __builtin_shufflevector(a0lo, a0hi,
          0, 1, 2, 3, 4, 5, 6, 7, 8, 9, 10, 11, 12, 13, 14, 15);
      const v8h a1lo = *(const v8h*)(ap1 + k0);
      const v8h a1hi = *(const v8h*)(ap1 + k0 + 16);
      const v16h a1 = __builtin_shufflevector(a1lo, a1hi,
          0, 1, 2, 3, 4, 5, 6, 7, 8, 9, 10, 11, 12, 13, 14, 15);
      #pragma unroll
      for (int j = 0; j < 4; ++j) {
        const _Float16* bp = lb + (j * 16 + row) * BROW + ks * 32 + hi * 8;
        const v8h blo = *(const v8h*)bp;
        const v8h bhi = *(const v8h*)(bp + 16);
        const v16h b = __builtin_shufflevector(blo, bhi,
            0, 1, 2, 3, 4, 5, 6, 7, 8, 9, 10, 11, 12, 13, 14, 15);
        acc[0][j] = __builtin_amdgcn_wmma_f32_16x16x32_f16(
            false, a0, false, b, (short)0, acc[0][j], false, false);
        acc[1][j] = __builtin_amdgcn_wmma_f32_16x16x32_f16(
            false, a1, false, b, (short)0, acc[1][j], false, false);
      }
    }

    if (uwave == 0) __builtin_amdgcn_s_wait_tensorcnt(0);
    __syncthreads();
  }

  // C/D layout: VGPR r, lanes 0-15 -> (M=r, N=lane); lanes 16-31 -> (M=8+r).
  #pragma unroll
  for (int i = 0; i < 2; ++i) {
    #pragma unroll
    for (int j = 0; j < 4; ++j) {
      #pragma unroll
      for (int r = 0; r < 8; ++r) {
        const int m = m0 + i * 16 + r + hi * 8;
        const int n = n0 + j * 16 + row;
        const float acv = acc[i][j][r];
        if constexpr (MODE <= 2) {
          const float val = acv + bias[n];
          const int b = m >> 9, s = m & (S - 1);
          const int h = n >> 6, d = n & (HD - 1);
          const long bh = (long)b * NH + h;
          if constexpr (MODE == 0) {
            ((_Float16*)out0)[(bh * S + s) * HD + d] = (_Float16)val;
          } else if constexpr (MODE == 1) {
            ((_Float16*)out0)[(bh * KP + (L + s)) * HD + d] = (_Float16)val;
          } else {
            const _Float16 hv = (_Float16)val;
            ((_Float16*)out0)[(bh * KP + (L + s)) * HD + d] = hv;
            ((_Float16*)out1)[(bh * HD + d) * KP + (L + s)] = hv;
          }
        } else if constexpr (MODE == 3) {
          const int bh = blockIdx.z;
          const int b = bh / NH;
          float val;
          if (n < KV) {
            const float mv = (n < L) ? 1.0f : amask[(long)b * S + (n - L)];
            val = acv * 0.125f + mv;       // 1/sqrt(64) = 0.125
          } else {
            val = -__builtin_inff();       // padded key -> prob exactly 0
          }
          ((_Float16*)out0)[((long)bh * S + m) * KP + n] = (_Float16)val;
        } else if constexpr (MODE == 4) {
          const int bh = blockIdx.z;
          const int b = bh / NH, h = bh % NH;
          ((_Float16*)out0)[((long)b * S + m) * H + h * HD + n] = (_Float16)acv;
        } else if constexpr (MODE == 5 || MODE == 7) {
          ((float*)out0)[(long)m * H + n] = acv + bias[n];
        } else {  // MODE == 6: exact GELU
          const float x = acv + bias[n];
          const float g = 0.5f * x * (1.0f + erff(x * 0.70710678118654752f));
          ((_Float16*)out0)[(long)m * FF + n] = (_Float16)g;
        }
      }
    }
  }
}

// ---------------------------------------------------------------------------
// wave-per-row softmax, in place over f16 rows of width KP (576 = 18*32)
// ---------------------------------------------------------------------------
__global__ __launch_bounds__(32)
void softmax_inplace(_Float16* __restrict__ sc) {
  using namespace cfg;
  _Float16* p = sc + (long)blockIdx.x * KP;
  const int lane = threadIdx.x;
  float v[18];
  float mx = -__builtin_inff();
  #pragma unroll
  for (int i = 0; i < 18; ++i) {
    v[i] = (float)p[lane + 32 * i];
    mx = fmaxf(mx, v[i]);
  }
  #pragma unroll
  for (int o = 16; o; o >>= 1) mx = fmaxf(mx, __shfl_xor(mx, o, 32));
  float s = 0.f;
  #pragma unroll
  for (int i = 0; i < 18; ++i) { v[i] = __expf(v[i] - mx); s += v[i]; }
  #pragma unroll
  for (int o = 16; o; o >>= 1) s += __shfl_xor(s, o, 32);
  const float inv = 1.0f / s;
  #pragma unroll
  for (int i = 0; i < 18; ++i) p[lane + 32 * i] = (_Float16)(v[i] * inv);
}

// ---------------------------------------------------------------------------
// fused residual + LayerNorm, block-per-row (H = 768 = 3*256)
// ---------------------------------------------------------------------------
__device__ inline float blk_sum(float v, float* red) {
  #pragma unroll
  for (int o = 16; o; o >>= 1) v += __shfl_xor(v, o, 32);
  const int w = threadIdx.x >> 5;
  __syncthreads();
  if ((threadIdx.x & 31) == 0) red[w] = v;
  __syncthreads();
  float t = 0.f;
  #pragma unroll
  for (int i = 0; i < 8; ++i) t += red[i];
  return t;
}

__global__ __launch_bounds__(256)
void ln_fused(const float* __restrict__ proj, const float* __restrict__ resid,
              const float* __restrict__ gam, const float* __restrict__ bet,
              float* __restrict__ out32, _Float16* __restrict__ out16) {
  using namespace cfg;
  __shared__ float red[8];
  const long rowb = (long)blockIdx.x * H;
  const int tid = threadIdx.x;
  float x[3];
  float s = 0.f;
  #pragma unroll
  for (int i = 0; i < 3; ++i) {
    x[i] = proj[rowb + tid + i * 256] + resid[rowb + tid + i * 256];
    s += x[i];
  }
  const float mu = blk_sum(s, red) * (1.0f / H);
  float vs = 0.f;
  #pragma unroll
  for (int i = 0; i < 3; ++i) { x[i] -= mu; vs += x[i] * x[i]; }
  const float var = blk_sum(vs, red) * (1.0f / H);
  const float inv = rsqrtf(var + 1e-12f);
  #pragma unroll
  for (int i = 0; i < 3; ++i) {
    const int c = tid + i * 256;
    const float y = x[i] * inv * gam[c] + bet[c];
    out32[rowb + c] = y;
    if (out16) out16[rowb + c] = (_Float16)y;
  }
}

// ---------------------------------------------------------------------------
extern "C" void kernel_launch(void* const* d_in, const int* in_sizes, int n_in,
                              void* d_out, int out_size, void* d_ws, size_t ws_size,
                              hipStream_t stream) {
  using namespace cfg;
  (void)in_sizes; (void)n_in; (void)out_size; (void)ws_size;

  const float* hidden = (const float*)d_in[0];
  const float* amask  = (const float*)d_in[1];
  const float* pastk  = (const float*)d_in[2];
  // d_in[3] past_value intentionally unused (reference bug uses past_key twice)
  const float* Wq = (const float*)d_in[4];  const float* bq = (const float*)d_in[5];
  const float* Wk = (const float*)d_in[6];  const float* bk = (const float*)d_in[7];
  const float* Wv = (const float*)d_in[8];  const float* bv = (const float*)d_in[9];
  const float* Wo = (const float*)d_in[10]; const float* bo = (const float*)d_in[11];
  const float* g1 = (const float*)d_in[12]; const float* b1 = (const float*)d_in[13];
  const float* Wi = (const float*)d_in[14]; const float* bi = (const float*)d_in[15];
  const float* Wf = (const float*)d_in[16]; const float* bf = (const float*)d_in[17];
  const float* g2 = (const float*)d_in[18]; const float* b2 = (const float*)d_in[19];

  char* base = (char*)d_ws;
  size_t off = 0;
  auto alloc = [&](size_t bytes) -> void* {
    void* p = base + off;
    off = (off + bytes + 255) & ~(size_t)255;
    return p;
  };

  _Float16* hs16  = (_Float16*)alloc((size_t)MTOK * H * 2);
  _Float16* WqT   = (_Float16*)alloc((size_t)H * H * 2);
  _Float16* WkT   = (_Float16*)alloc((size_t)H * H * 2);
  _Float16* WvT   = (_Float16*)alloc((size_t)H * H * 2);
  _Float16* WoT   = (_Float16*)alloc((size_t)H * H * 2);
  _Float16* WiT   = (_Float16*)alloc((size_t)FF * H * 2);
  _Float16* WfT   = (_Float16*)alloc((size_t)H * FF * 2);
  _Float16* q16   = (_Float16*)alloc((size_t)B * NH * S * HD * 2);
  _Float16* k16   = (_Float16*)alloc((size_t)B * NH * KP * HD * 2);
  _Float16* v16   = (_Float16*)alloc((size_t)B * NH * KP * HD * 2);
  _Float16* vT16  = (_Float16*)alloc((size_t)B * NH * HD * KP * 2);
  _Float16* sc16  = (_Float16*)alloc((size_t)B * NH * S * KP * 2);
  _Float16* ctx16 = (_Float16*)alloc((size_t)MTOK * H * 2);
  float*    proj32 = (float*)alloc((size_t)MTOK * H * 4);   // reused by Wo and Wf
  float*    attn32 = (float*)alloc((size_t)MTOK * H * 4);
  _Float16* attn16 = (_Float16*)alloc((size_t)MTOK * H * 2);
  _Float16* inter16 = (_Float16*)alloc((size_t)MTOK * FF * 2);

  // --- precision / layout pre-passes ---
  cvt_f32_f16<<<4096, 256, 0, stream>>>(hidden, hs16, MTOK * H);
  transpose_f32_to_f16<<<dim3(H / 32, H / 32), 256, 0, stream>>>(Wq, WqT, H, H);
  transpose_f32_to_f16<<<dim3(H / 32, H / 32), 256, 0, stream>>>(Wk, WkT, H, H);
  transpose_f32_to_f16<<<dim3(H / 32, H / 32), 256, 0, stream>>>(Wv, WvT, H, H);
  transpose_f32_to_f16<<<dim3(H / 32, H / 32), 256, 0, stream>>>(Wo, WoT, H, H);
  transpose_f32_to_f16<<<dim3(FF / 32, H / 32), 256, 0, stream>>>(Wi, WiT, H, FF);
  transpose_f32_to_f16<<<dim3(H / 32, FF / 32), 256, 0, stream>>>(Wf, WfT, FF, H);
  past_to_kv<<<(B * NH * L * HD + 255) / 256, 256, 0, stream>>>(pastk, k16, v16, vT16);
  zero_pads<<<(B * NH * (KP - KV) * HD + 255) / 256, 256, 0, stream>>>(k16, v16, vT16);

  // --- QKV projections (8192x768x768 each) ---
  const dim3 gproj(H / 64, (int)(MTOK / 256));
  gemm_wmma<0><<<gproj, 256, 0, stream>>>(hs16, WqT, bq, q16, nullptr, nullptr,
                                          (int)MTOK, H, H, H, H);
  gemm_wmma<1><<<gproj, 256, 0, stream>>>(hs16, WkT, bk, k16, nullptr, nullptr,
                                          (int)MTOK, H, H, H, H);
  gemm_wmma<2><<<gproj, 256, 0, stream>>>(hs16, WvT, bv, v16, vT16, nullptr,
                                          (int)MTOK, H, H, H, H);

  // --- attention scores: per (b,h) 512x576x64, scale+mask in epilogue ---
  const dim3 gsc(KP / 64, S / 256, B * NH);
  gemm_wmma<3><<<gsc, 256, 0, stream>>>(q16, k16, nullptr, sc16, nullptr, amask,
                                        S, KP, HD, HD, HD);

  softmax_inplace<<<B * NH * S, 32, 0, stream>>>(sc16);

  // --- context: per (b,h) 512x64x576, heads merged in epilogue ---
  const dim3 gctx(HD / 64, S / 256, B * NH);
  gemm_wmma<4><<<gctx, 256, 0, stream>>>(sc16, vT16, nullptr, ctx16, nullptr, nullptr,
                                         S, HD, KP, KP, KP);

  // --- output projection + residual LN1 ---
  gemm_wmma<5><<<gproj, 256, 0, stream>>>(ctx16, WoT, bo, proj32, nullptr, nullptr,
                                          (int)MTOK, H, H, H, H);
  ln_fused<<<(int)MTOK, 256, 0, stream>>>(proj32, hidden, g1, b1, attn32, attn16);

  // --- FFN: Wi + GELU, then Wf + residual LN2 -> d_out ---
  const dim3 gff1(FF / 64, (int)(MTOK / 256));
  gemm_wmma<6><<<gff1, 256, 0, stream>>>(attn16, WiT, bi, inter16, nullptr, nullptr,
                                         (int)MTOK, FF, H, H, H);
  const dim3 gff2(H / 64, (int)(MTOK / 256));
  gemm_wmma<7><<<gff2, 256, 0, stream>>>(inter16, WfT, bf, proj32, nullptr, nullptr,
                                         (int)MTOK, H, FF, FF, FF);
  ln_fused<<<(int)MTOK, 256, 0, stream>>>(proj32, attn32, g2, b2, (float*)d_out, nullptr);
}
